// AxCoreLinearFP16_43602507989105
// MI455X (gfx1250) — matrix operand order repaired and checked by reading, not compiled, not run on hardware
//
#include <hip/hip_runtime.h>

typedef _Float16 v16h __attribute__((ext_vector_type(16)));
typedef _Float16 v8h  __attribute__((ext_vector_type(8)));
typedef float    v8f  __attribute__((ext_vector_type(8)));

#define BM 128
#define BN 128
#define BK 64            // halves per K chunk; one 128-wide scale group == 2 chunks
#define LDS_STRIDE 72    // halves per LDS row: 64 data + 8 pad (144B, 16B-aligned, conflict-free)
#define QGROUP 128

#define TILE_HALVES (BM * LDS_STRIDE)
#define BUF_HALVES  (2 * TILE_HALVES)
#define LDS_BYTES   (2 * BUF_HALVES * 2)   // double buffered: 73728 B (gfx1250: 320KB/WGP)

// CDNA5 async global->LDS copy, 16 bytes per lane. Flat pointers into the LDS
// aperture carry the LDS byte address in their low 32 bits (ISA 10.2), so
// truncation gives the ds address for the VDST operand.
__device__ __forceinline__ void async_copy_b128(const _Float16* gptr, _Float16* lptr) {
    unsigned lds_addr = (unsigned)(unsigned long long)(void*)lptr;
    asm volatile("global_load_async_to_lds_b128 %0, %1, off"
                 :: "v"(lds_addr), "v"(gptr)
                 : "memory");
}

__device__ __forceinline__ void wait_async_le8() {
    asm volatile("s_wait_asynccnt 0x8" ::: "memory");
}
__device__ __forceinline__ void wait_async_0() {
    asm volatile("s_wait_asynccnt 0x0" ::: "memory");
}

// 8 async b128 copies per thread per K chunk (4 segs X tile + 4 segs W tile).
__device__ __forceinline__ void prefetch_chunk(const _Float16* __restrict__ X,
                                               const _Float16* __restrict__ W,
                                               _Float16* Xs, _Float16* Ws,
                                               int m0, int n0, int k0, int K, int tid) {
#pragma unroll
    for (int i = 0; i < 4; ++i) {
        int s   = tid + i * 256;     // 1024 segments of 8 halves = 128 rows x 64 halves
        int row = s >> 3;
        int cs  = (s & 7) * 8;
        async_copy_b128(X + (long long)(m0 + row) * K + k0 + cs,
                        Xs + row * LDS_STRIDE + cs);
    }
#pragma unroll
    for (int i = 0; i < 4; ++i) {
        int s   = tid + i * 256;
        int row = s >> 3;
        int cs  = (s & 7) * 8;
        async_copy_b128(W + (long long)(n0 + row) * K + k0 + cs,
                        Ws + row * LDS_STRIDE + cs);
    }
}

__global__ __launch_bounds__(256, 1)
void AxCoreLinearFP16_kernel(const _Float16* __restrict__ X,   // [M, K]
                             const _Float16* __restrict__ W,   // [N, K]
                             const _Float16* __restrict__ S,   // [N, K/128]
                             _Float16* __restrict__ Y,         // [M, N]
                             int M, int N, int K) {
    extern __shared__ _Float16 lds[];

    const int tid  = threadIdx.x;
    const int lane = tid & 31;
    const int wave = tid >> 5;
    const int wm   = wave >> 2;          // 0..1 : wave row (64 M-rows)
    const int wn   = wave & 3;           // 0..3 : wave col (32 N-cols)
    const int l15  = lane & 15;
    const int lhi  = lane >> 4;

    const int m0 = blockIdx.y * BM;
    const int n0 = blockIdx.x * BN;

    const int nchunk  = K / BK;
    const int ngroups = K / QGROUP;

    v8f acc[4][2];
#pragma unroll
    for (int mt = 0; mt < 4; ++mt)
#pragma unroll
        for (int nt = 0; nt < 2; ++nt)
#pragma unroll
            for (int i = 0; i < 8; ++i) acc[mt][nt][i] = 0.0f;

    // Prefetch chunk 0 into buffer 0.
    prefetch_chunk(X, W, lds, lds + TILE_HALVES, m0, n0, 0, K, tid);

    for (int gi = 0; gi < ngroups; ++gi) {
        // Dequant scale: per (N column, 128-group). In the C/D layout lane%16 is
        // the N column, so it is one scalar per lane per 16-wide N tile.
        const float s0f = (float)S[(long long)(n0 + wn * 32 + l15)      * ngroups + gi];
        const float s1f = (float)S[(long long)(n0 + wn * 32 + 16 + l15) * ngroups + gi];

        v8f gacc[4][2];   // unscaled partial sums for this group (C=0 on first WMMA)

#pragma unroll
        for (int half = 0; half < 2; ++half) {   // chunk c = 2*gi + half
            const int c   = gi * 2 + half;
            const int buf = c & 1;
            if (c + 1 < nchunk) {
                _Float16* nXs = lds + (buf ^ 1) * BUF_HALVES;
                prefetch_chunk(X, W, nXs, nXs + TILE_HALVES, m0, n0, (c + 1) * BK, K, tid);
                wait_async_le8();        // chunk c complete; c+1 still in flight
            } else {
                wait_async_0();
            }
            __syncthreads();

            const _Float16* Xs = lds + buf * BUF_HALVES;
            const _Float16* Ws = Xs + TILE_HALVES;

#pragma unroll
            for (int ks = 0; ks < 2; ++ks) {     // two 16x16x32 K-steps per chunk
                // A frags: lanes 0-15 K {0..7,16..23}; lanes 16-31 K {8..15,24..31}
                v16h a[4];
#pragma unroll
                for (int mt = 0; mt < 4; ++mt) {
                    int r    = wm * 64 + mt * 16 + l15;
                    int base = r * LDS_STRIDE + ks * 32 + (lhi ? 8 : 0);
                    v8h lo = *(const v8h*)(Xs + base);
                    v8h hi = *(const v8h*)(Xs + base + 16);
#pragma unroll
                    for (int i = 0; i < 8; ++i) { a[mt][i] = lo[i]; a[mt][i + 8] = hi[i]; }
                }
                // B frags: lane holds W row (= B column); lanes 0-15 K {0..7,16..23}; lanes 16-31 K {8..15,24..31}
                // No scaling here -- WMMA consumes the ds_load result directly.
                v16h b[2];
#pragma unroll
                for (int nt = 0; nt < 2; ++nt) {
                    int r    = wn * 32 + nt * 16 + l15;
                    int base = r * LDS_STRIDE + ks * 32 + (lhi ? 8 : 0);
                    v8h lo = *(const v8h*)(Ws + base);
                    v8h hi = *(const v8h*)(Ws + base + 16);
#pragma unroll
                    for (int i = 0; i < 8; ++i) { b[nt][i] = lo[i]; b[nt][i + 8] = hi[i]; }
                }
#pragma unroll
                for (int mt = 0; mt < 4; ++mt)
#pragma unroll
                    for (int nt = 0; nt < 2; ++nt) {
                        if (half == 0 && ks == 0) {
                            v8f z;
#pragma unroll
                            for (int i = 0; i < 8; ++i) z[i] = 0.0f;
                            gacc[mt][nt] = __builtin_amdgcn_wmma_f32_16x16x32_f16(
                                false, a[mt], false, b[nt], (short)0, z, false, false);
                        } else {
                            gacc[mt][nt] = __builtin_amdgcn_wmma_f32_16x16x32_f16(
                                false, a[mt], false, b[nt], (short)0, gacc[mt][nt], false, false);
                        }
                    }
            }
            __syncthreads();   // buffer c&1 must drain before next prefetch overwrites it
        }

        // Fold dequant scale once per group: 64 v_fmac_f32 per lane per group.
#pragma unroll
        for (int mt = 0; mt < 4; ++mt)
#pragma unroll
            for (int nt = 0; nt < 2; ++nt) {
                float s = nt ? s1f : s0f;
#pragma unroll
                for (int i = 0; i < 8; ++i)
                    acc[mt][nt][i] += s * gacc[mt][nt][i];
            }
    }

    // Epilogue: VGPR r holds M row r (+8 for lanes 16-31), lane%16 = N column.
#pragma unroll
    for (int mt = 0; mt < 4; ++mt) {
#pragma unroll
        for (int nt = 0; nt < 2; ++nt) {
            int col     = n0 + wn * 32 + nt * 16 + l15;
            int rowbase = m0 + wm * 64 + mt * 16 + (lhi ? 8 : 0);
#pragma unroll
            for (int r = 0; r < 8; ++r) {
                Y[(long long)(rowbase + r) * N + col] = (_Float16)acc[mt][nt][r];
            }
        }
    }
}

extern "C" void kernel_launch(void* const* d_in, const int* in_sizes, int n_in,
                              void* d_out, int out_size, void* d_ws, size_t ws_size,
                              hipStream_t stream) {
    (void)n_in; (void)d_ws; (void)ws_size; (void)out_size;
    const _Float16* X = (const _Float16*)d_in[0];
    const _Float16* W = (const _Float16*)d_in[1];
    const _Float16* S = (const _Float16*)d_in[2];
    _Float16*       Y = (_Float16*)d_out;

    // Recover dims: |W| = N*K, |scales| = N*(K/128) => K, N, then M from |x|.
    const long long wsz = in_sizes[1];
    const long long ssz = in_sizes[2];
    const int K = (int)(128LL * wsz / ssz);
    const int N = (int)(wsz / K);
    const int M = (int)((long long)in_sizes[0] / K);

    dim3 grid(N / BN, M / BM);
    AxCoreLinearFP16_kernel<<<grid, 256, LDS_BYTES, stream>>>(X, W, S, Y, M, N, K);
}
